// Mamba2Simple_40243843564249
// MI455X (gfx1250) — compile-verified, matrix operand order checked by reading, not run on hardware
//
#include <hip/hip_runtime.h>
#include <stdint.h>

// ---------------- problem constants ----------------
#define D_MODEL   2048
#define D_INNER   4096
#define D_STATE   128
#define HEADDIM   64
#define NHEADS    64
#define D_XBC     4352
#define D_IN_PROJ 8512
#define CHUNK     128
#define NCHUNK    16
#define BATCH     2
#define SEQ       2048
#define MTOT      4096   // BATCH*SEQ

// ---------------- types ----------------
typedef unsigned short u16;
typedef u16    u16x8   __attribute__((ext_vector_type(8)));
typedef u16    u16x16  __attribute__((ext_vector_type(16)));
typedef __bf16 bf16x16 __attribute__((ext_vector_type(16)));
typedef float  floatx8 __attribute__((ext_vector_type(8)));
typedef int    ivec4   __attribute__((vector_size(16)));   // matches builtin param type

// ---------------- gfx1250 async copy (GLOBAL_LOAD_ASYNC_TO_LDS_B128) ----------------
#if __has_builtin(__builtin_amdgcn_global_load_async_to_lds_b128) && \
    __has_builtin(__builtin_amdgcn_s_wait_asynccnt)
#define USE_ASYNC_COPY 1
#else
#define USE_ASYNC_COPY 0
#endif

__device__ __forceinline__ void async_ld_b128(u16* lds_dst, const u16* gsrc) {
#if USE_ASYNC_COPY
  __builtin_amdgcn_global_load_async_to_lds_b128(
      (__attribute__((address_space(1))) ivec4*)gsrc,
      (__attribute__((address_space(3))) ivec4*)lds_dst,
      0, 0);
#else
  *(u16x8*)lds_dst = *(const u16x8*)gsrc;   // sync fallback: global->vgpr->lds
#endif
}

#if USE_ASYNC_COPY
#define WAIT_ASYNC(n) __builtin_amdgcn_s_wait_asynccnt(n)
#else
#define WAIT_ASYNC(n) ((void)0)
#endif

__device__ __forceinline__ u16 f2bf(float f) {
  unsigned x = __float_as_uint(f);
  unsigned r = x + 0x7FFFu + ((x >> 16) & 1u);   // round-to-nearest-even
  return (u16)(r >> 16);
}

__device__ __forceinline__ floatx8 zacc() {
  floatx8 z = {0.f, 0.f, 0.f, 0.f, 0.f, 0.f, 0.f, 0.f};
  return z;
}

__device__ __forceinline__ floatx8 wmma_bf16(bf16x16 a, bf16x16 b, floatx8 c) {
  // D = A x B + C, 16x16x32, bf16 in / f32 accumulate
  return __builtin_amdgcn_wmma_f32_16x16x32_bf16(false, a, false, b, (short)0, c,
                                                 false, false);
}

// Load one 16x32 bf16 fragment (A or B operand) from an LDS tile stored
// row-major [rows][ld] halfwords.  Per ISA 7.12.2 layout:
//   lanes 0-15 : row = row0+lane,    K = {0..7, 16..23}   (+k0)
//   lanes16-31 : row = row0+lane-16, K = {8..15, 24..31}  (+k0)
// -> two contiguous 16-byte ds reads per lane.
__device__ __forceinline__ bf16x16 frag_ld(const u16* base, int ld, int row0, int k0) {
  int lane = threadIdx.x & 31;
  const u16* p = base + (size_t)(row0 + (lane & 15)) * ld + k0 + ((lane >> 4) << 3);
  u16x8 lo = *(const u16x8*)(p);
  u16x8 hi = *(const u16x8*)(p + 16);
  u16x16 t = __builtin_shufflevector(lo, hi, 0, 1, 2, 3, 4, 5, 6, 7,
                                     8, 9, 10, 11, 12, 13, 14, 15);
  return __builtin_bit_cast(bf16x16, t);
}

// ---------------- fp32 -> bf16 convert ----------------
__global__ __launch_bounds__(256)
void f2bf_kernel(const float* __restrict__ in, u16* __restrict__ out, int n) {
  int i = blockIdx.x * 256 + threadIdx.x;
  if (i < n) out[i] = f2bf(in[i]);
}

// ---------------- GEMM: C[M,N] = A[M,K] * B[N,K]^T ----------------
// block 128x128, 8 waves, wave tile 64x32 (4x2 WMMA frags), K step 64
// (2 WMMA k-slices per phase -> 16 WMMAs between barriers).
// Double-buffered LDS; tiles stream in via async-to-LDS, overlapping WMMA.
#define GBM 128
#define GBN 128
#define GBK 64
#define GKPAD 72   // 64 + 8 halfwords -> 144B row stride, conflict-free b128 reads

__global__ __launch_bounds__(256)
void gemm_bf16_abt(const u16* __restrict__ A, const u16* __restrict__ B,
                   float* __restrict__ C, int M, int N, int K) {
  __shared__ u16 As[2][GBM][GKPAD];
  __shared__ u16 Bs[2][GBN][GKPAD];
  const int tid  = threadIdx.x;
  const int lane = tid & 31;
  const int wave = tid >> 5;
  const int wm   = wave >> 2;          // 0..1
  const int wn   = wave & 3;           // 0..3
  const int lh   = (lane >> 4) << 3;   // 0 or 8
  const int lm   = lane & 15;
  const int bm   = blockIdx.y * GBM;
  const int bn   = blockIdx.x * GBN;

  floatx8 acc[4][2];
#pragma unroll
  for (int i = 0; i < 4; ++i)
#pragma unroll
    for (int j = 0; j < 2; ++j) acc[i][j] = zacc();

  // copy mapping: 128 rows x 64 cols bf16 = 128B/row = 8 x b128 per row;
  // 256 threads x 4 chunks: chunk q = tid + 256*i -> row q>>3, col (q&7)*8
  const int ldr = tid >> 3;            // base row (0..31), +32 per i
  const int ldc = (tid & 7) << 3;      // halfword col: 0,8,...,56
  // Edge handling: clamp out-of-range B rows to N-1 instead of zero-filling.
  // Garbage in those LDS rows only reaches output columns >= N, which the
  // epilogue never stores -> branch-free, maskless copy path.
  int bnr[4];
#pragma unroll
  for (int i = 0; i < 4; ++i) {
    int r = bn + ldr + 32 * i;
    bnr[i] = (r >= N) ? (N - 1) : r;
  }

  auto issue_tile = [&](int buf, int k0) {
#pragma unroll
    for (int i = 0; i < 4; ++i) {
      int r = ldr + 32 * i;
      async_ld_b128(&As[buf][r][ldc], A + (size_t)(bm + r) * K + k0 + ldc);
      async_ld_b128(&Bs[buf][r][ldc], B + (size_t)bnr[i] * K + k0 + ldc);
    }
  };

  issue_tile(0, 0);
  int kb = 0;
  for (int k0 = 0; k0 < K; k0 += GBK) {
    if (k0 + GBK < K) {
      issue_tile(kb ^ 1, k0 + GBK);  // overlap next tile's copies with compute
      WAIT_ASYNC(8);                 // in-order completion: first batch landed
    } else {
      WAIT_ASYNC(0);
    }
    __syncthreads();                 // all waves' tile data visible

#pragma unroll
    for (int ks = 0; ks < 2; ++ks) {
      bf16x16 af[4], bfv[2];
#pragma unroll
      for (int tm = 0; tm < 4; ++tm)
        af[tm] = frag_ld(&As[kb][0][0], GKPAD, wm * 64 + tm * 16, ks * 32);
#pragma unroll
      for (int tn = 0; tn < 2; ++tn)
        bfv[tn] = frag_ld(&Bs[kb][0][0], GKPAD, wn * 32 + tn * 16, ks * 32);
#pragma unroll
      for (int tm = 0; tm < 4; ++tm)
#pragma unroll
        for (int tn = 0; tn < 2; ++tn)
          acc[tm][tn] = wmma_bf16(af[tm], bfv[tn], acc[tm][tn]);
    }

    __syncthreads();                 // done reading buf kb; it may be refilled
    kb ^= 1;
  }

  // D layout: VGPR j -> row = +j (+8 for upper half-lanes), col = lane&15
#pragma unroll
  for (int tm = 0; tm < 4; ++tm)
#pragma unroll
    for (int tn = 0; tn < 2; ++tn)
#pragma unroll
      for (int j = 0; j < 8; ++j) {
        int row = bm + wm * 64 + tm * 16 + lh + j;
        int col = bn + wn * 32 + tn * 16 + lm;
        if (col < N) C[(size_t)row * N + col] = acc[tm][tn][j];
      }
}

// ---------------- depthwise causal conv (width 4) + bias + SiLU ----------------
__global__ __launch_bounds__(256)
void conv_silu_kernel(const float* __restrict__ zx, const float* __restrict__ conv_w,
                      const float* __restrict__ conv_b, float* __restrict__ xbc) {
  size_t idx = (size_t)blockIdx.x * 256 + threadIdx.x;
  const size_t total = (size_t)MTOT * D_XBC;
  if (idx >= total) return;
  int row = (int)(idx / D_XBC);
  int ch  = (int)(idx % D_XBC);
  int t   = row & (SEQ - 1);
  float a = conv_b[ch];
#pragma unroll
  for (int k = 0; k < 4; ++k) {
    int tt = t - 3 + k;
    if (tt >= 0)
      a += conv_w[ch * 4 + k] * zx[(size_t)(row - 3 + k) * D_IN_PROJ + D_INNER + ch];
  }
  float sig = 1.f / (1.f + __expf(-a));
  xbc[idx] = a * sig;
}

// ---------------- SSD per-chunk kernel (one block per (b,c,h)) ----------------
// Everything for a 128-token chunk lives in LDS (~171KB of the 320KB WGP LDS):
//   G = C.B^T ; GL = tril(G*exp(acs_l - acs_s)) ; Y_diag = GL.xdt
//   states = (x*dt*decay)^T . B
#define LPAD 136   // 128 + 8 halfwords -> 272B stride, conflict-free

__global__ __launch_bounds__(256)
void ssd_chunk_kernel(const float* __restrict__ zx, const float* __restrict__ xbc,
                      const float* __restrict__ dt_bias, const float* __restrict__ A_log,
                      float* __restrict__ y, float* __restrict__ states,
                      float* __restrict__ acs_out, float* __restrict__ cdecay_out) {
  __shared__ u16 Cs[CHUNK][LPAD];
  __shared__ u16 Bs[CHUNK][LPAD];
  __shared__ u16 Bt[D_STATE][LPAD];
  __shared__ u16 GL[CHUNK][LPAD];
  __shared__ u16 xdtT[HEADDIM][LPAD];
  __shared__ u16 xdecT[HEADDIM][LPAD];
  __shared__ float dtv[CHUNK];
  __shared__ float acs[CHUNK];

  const int c = blockIdx.x, h = blockIdx.y, b = blockIdx.z;
  const int tid  = threadIdx.x;
  const int lane = tid & 31, wave = tid >> 5;
  const int lh = (lane >> 4) << 3, lm = lane & 15;
  const int t0 = c * CHUNK;
  const size_t rowbase = (size_t)(b * SEQ + t0);

  // --- stage 1: dt = softplus(dt_raw + bias); acs = cumsum(dt * A) ---
  float Ah = -__expf(A_log[h]);
  if (tid < CHUNK) {
    float draw = zx[(rowbase + tid) * D_IN_PROJ + (D_INNER + D_XBC) + h] + dt_bias[h];
    float d = (draw > 20.f) ? draw : log1pf(__expf(draw));
    dtv[tid] = d;
    acs[tid] = d * Ah;
  }
  __syncthreads();
  for (int off = 1; off < CHUNK; off <<= 1) {  // Hillis-Steele inclusive scan
    float v = 0.f;
    if (tid < CHUNK && tid >= off) v = acs[tid - off];
    __syncthreads();
    if (tid < CHUNK) acs[tid] += v;
    __syncthreads();
  }
  const float a_last = acs[CHUNK - 1];

  // --- stage 2: fill bf16 tiles (C, B, B^T, x*dt transposed, decayed) ---
#pragma unroll 2
  for (int i = 0; i < 16; ++i) {
    int v  = tid + 256 * i;
    int r  = v >> 5;
    int c4 = (v & 31) << 2;
    const float* rowp = xbc + (rowbase + r) * D_XBC;
    float4 cv = *(const float4*)(rowp + (D_INNER + D_STATE) + c4);
    float4 bv = *(const float4*)(rowp + D_INNER + c4);
    Cs[r][c4 + 0] = f2bf(cv.x); Cs[r][c4 + 1] = f2bf(cv.y);
    Cs[r][c4 + 2] = f2bf(cv.z); Cs[r][c4 + 3] = f2bf(cv.w);
    u16 b0 = f2bf(bv.x), b1 = f2bf(bv.y), b2 = f2bf(bv.z), b3 = f2bf(bv.w);
    Bs[r][c4 + 0] = b0; Bs[r][c4 + 1] = b1; Bs[r][c4 + 2] = b2; Bs[r][c4 + 3] = b3;
    Bt[c4 + 0][r] = b0; Bt[c4 + 1][r] = b1; Bt[c4 + 2][r] = b2; Bt[c4 + 3][r] = b3;
  }
#pragma unroll 2
  for (int i = 0; i < 8; ++i) {
    int v  = tid + 256 * i;
    int r  = v >> 4;
    int c4 = (v & 15) << 2;
    float4 xv = *(const float4*)(xbc + (rowbase + r) * D_XBC + h * HEADDIM + c4);
    float sdt  = dtv[r];
    float sdec = __expf(a_last - acs[r]);
    xdtT[c4 + 0][r]  = f2bf(xv.x * sdt);        xdtT[c4 + 1][r]  = f2bf(xv.y * sdt);
    xdtT[c4 + 2][r]  = f2bf(xv.z * sdt);        xdtT[c4 + 3][r]  = f2bf(xv.w * sdt);
    xdecT[c4 + 0][r] = f2bf(xv.x * sdt * sdec); xdecT[c4 + 1][r] = f2bf(xv.y * sdt * sdec);
    xdecT[c4 + 2][r] = f2bf(xv.z * sdt * sdec); xdecT[c4 + 3][r] = f2bf(xv.w * sdt * sdec);
  }
  __syncthreads();

  // --- stage 3: G = C . B^T (128x128, K=128); apply causal decay mask -> GL ---
  {
    bf16x16 cf[4];
#pragma unroll
    for (int ks = 0; ks < 4; ++ks) cf[ks] = frag_ld(&Cs[0][0], LPAD, wave * 16, ks * 32);
#pragma unroll
    for (int tn = 0; tn < 8; ++tn) {
      floatx8 av = zacc();
#pragma unroll
      for (int ks = 0; ks < 4; ++ks)
        av = wmma_bf16(cf[ks], frag_ld(&Bs[0][0], LPAD, tn * 16, ks * 32), av);
#pragma unroll
      for (int j = 0; j < 8; ++j) {
        int l = wave * 16 + lh + j;
        int s = tn * 16 + lm;
        float vv = av[j];
        vv = (l >= s) ? vv * __expf(acs[l] - acs[s]) : 0.f;
        GL[l][s] = f2bf(vv);
      }
    }
  }
  __syncthreads();

  // --- stage 4: Y_diag = GL . xdt (128x64, K=128) ---
  {
    bf16x16 gf[4];
#pragma unroll
    for (int ks = 0; ks < 4; ++ks) gf[ks] = frag_ld(&GL[0][0], LPAD, wave * 16, ks * 32);
#pragma unroll
    for (int tn = 0; tn < 4; ++tn) {
      floatx8 av = zacc();
#pragma unroll
      for (int ks = 0; ks < 4; ++ks)
        av = wmma_bf16(gf[ks], frag_ld(&xdtT[0][0], LPAD, tn * 16, ks * 32), av);
#pragma unroll
      for (int j = 0; j < 8; ++j) {
        int l = wave * 16 + lh + j;
        int p = tn * 16 + lm;
        y[(rowbase + l) * (size_t)D_INNER + h * HEADDIM + p] = av[j];
      }
    }
  }

  // --- stage 5: states[p,n] = sum_l xdec[p,l] * Bt[n,l]  (64x128, K=128) ---
  {
    int pm = wave & 3;
    int ng = (wave >> 2) * 64;
    bf16x16 xf[4];
#pragma unroll
    for (int ks = 0; ks < 4; ++ks) xf[ks] = frag_ld(&xdecT[0][0], LPAD, pm * 16, ks * 32);
#pragma unroll
    for (int it = 0; it < 4; ++it) {
      floatx8 av = zacc();
#pragma unroll
      for (int ks = 0; ks < 4; ++ks)
        av = wmma_bf16(xf[ks], frag_ld(&Bt[0][0], LPAD, ng + it * 16, ks * 32), av);
#pragma unroll
      for (int j = 0; j < 8; ++j) {
        int p = pm * 16 + lh + j;
        int n = ng + it * 16 + lm;
        states[((((size_t)b * NCHUNK + c) * NHEADS + h) * HEADDIM + p) * D_STATE + n] = av[j];
      }
    }
  }

  if (tid < CHUNK) acs_out[(((size_t)b * NHEADS + h) * NCHUNK + c) * CHUNK + tid] = acs[tid];
  if (tid == 0)    cdecay_out[((size_t)b * NHEADS + h) * NCHUNK + c] = __expf(acs[CHUNK - 1]);
}

// ---------------- inter-chunk recurrence (16 sequential steps per (b,h)) ----------------
__global__ __launch_bounds__(256)
void scan_kernel(const float* __restrict__ states, const float* __restrict__ cdecay,
                 float* __restrict__ prev) {
  const int bh = blockIdx.x;            // 0..127
  const int b = bh >> 6, h = bh & 63;
  const int tid = threadIdx.x;
  float carry[32];
#pragma unroll
  for (int i = 0; i < 32; ++i) carry[i] = 0.f;
  for (int c = 0; c < NCHUNK; ++c) {
    size_t base = (((size_t)b * NCHUNK + c) * NHEADS + h) * (size_t)(HEADDIM * D_STATE);
    float d = cdecay[((size_t)b * NHEADS + h) * NCHUNK + c];
#pragma unroll
    for (int i = 0; i < 32; ++i) {
      size_t e = base + tid + 256 * i;
      prev[e]  = carry[i];                       // state BEFORE this chunk
      carry[i] = carry[i] * d + states[e];
    }
  }
}

// ---------------- Y_off = (C * exp(acs)) . prev^T  + D * x, accumulated into y ----------------
__global__ __launch_bounds__(256)
void ssd_off_kernel(const float* __restrict__ xbc, const float* __restrict__ prev,
                    const float* __restrict__ acs_in, const float* __restrict__ Dvec,
                    float* __restrict__ y) {
  __shared__ u16 Ce[CHUNK][LPAD];
  __shared__ u16 Pv[HEADDIM][LPAD];
  __shared__ float acs[CHUNK];
  const int c = blockIdx.x, h = blockIdx.y, b = blockIdx.z;
  const int tid = threadIdx.x;
  const int lane = tid & 31, wave = tid >> 5;
  const int lh = (lane >> 4) << 3, lm = lane & 15;
  const int t0 = c * CHUNK;
  const size_t rowbase = (size_t)(b * SEQ + t0);

  if (tid < CHUNK) acs[tid] = acs_in[(((size_t)b * NHEADS + h) * NCHUNK + c) * CHUNK + tid];
  __syncthreads();

#pragma unroll 2
  for (int i = 0; i < 16; ++i) {
    int v = tid + 256 * i;
    int r = v >> 5;
    int c4 = (v & 31) << 2;
    float4 cv = *(const float4*)(xbc + (rowbase + r) * D_XBC + (D_INNER + D_STATE) + c4);
    float e = __expf(acs[r]);
    Ce[r][c4 + 0] = f2bf(cv.x * e); Ce[r][c4 + 1] = f2bf(cv.y * e);
    Ce[r][c4 + 2] = f2bf(cv.z * e); Ce[r][c4 + 3] = f2bf(cv.w * e);
  }
  size_t pbase = (((size_t)b * NCHUNK + c) * NHEADS + h) * (size_t)(HEADDIM * D_STATE);
#pragma unroll 2
  for (int i = 0; i < 8; ++i) {
    int v = tid + 256 * i;
    int p = v >> 5;
    int n4 = (v & 31) << 2;
    float4 pv = *(const float4*)(prev + pbase + (size_t)p * D_STATE + n4);
    Pv[p][n4 + 0] = f2bf(pv.x); Pv[p][n4 + 1] = f2bf(pv.y);
    Pv[p][n4 + 2] = f2bf(pv.z); Pv[p][n4 + 3] = f2bf(pv.w);
  }
  __syncthreads();

  bf16x16 cf[4];
#pragma unroll
  for (int ks = 0; ks < 4; ++ks) cf[ks] = frag_ld(&Ce[0][0], LPAD, wave * 16, ks * 32);
  float Dh = Dvec[h];
#pragma unroll
  for (int tn = 0; tn < 4; ++tn) {
    floatx8 av = zacc();
#pragma unroll
    for (int ks = 0; ks < 4; ++ks)
      av = wmma_bf16(cf[ks], frag_ld(&Pv[0][0], LPAD, tn * 16, ks * 32), av);
#pragma unroll
    for (int j = 0; j < 8; ++j) {
      int l = wave * 16 + lh + j;
      int p = tn * 16 + lm;
      size_t yi = (rowbase + l) * (size_t)D_INNER + h * HEADDIM + p;
      float xval = xbc[(rowbase + l) * D_XBC + h * HEADDIM + p];
      y[yi] += av[j] + xval * Dh;
    }
  }
}

// ---------------- gated RMSNorm -> bf16 for the output GEMM ----------------
__global__ __launch_bounds__(256)
void rmsnorm_gate_kernel(const float* __restrict__ y, const float* __restrict__ zx,
                         const float* __restrict__ normw, u16* __restrict__ ybf) {
  __shared__ float buf[D_INNER];
  __shared__ float red[256];
  const int r = blockIdx.x;
  const int tid = threadIdx.x;
  const float* yr = y + (size_t)r * D_INNER;
  const float* zr = zx + (size_t)r * D_IN_PROJ;   // z = first D_INNER cols
  float s = 0.f;
  for (int i = tid; i < D_INNER; i += 256) {
    float z = zr[i];
    float yg = yr[i] * z / (1.f + __expf(-z));    // y * silu(z)
    buf[i] = yg;
    s += yg * yg;
  }
  red[tid] = s;
  __syncthreads();
  for (int o = 128; o > 0; o >>= 1) {
    if (tid < o) red[tid] += red[tid + o];
    __syncthreads();
  }
  float scale = rsqrtf(red[0] / (float)D_INNER + 1e-5f);
  for (int i = tid; i < D_INNER; i += 256)
    ybf[(size_t)r * D_INNER + i] = f2bf(buf[i] * scale * normw[i]);
}

// ---------------- host-side launch ----------------
extern "C" void kernel_launch(void* const* d_in, const int* in_sizes, int n_in,
                              void* d_out, int out_size, void* d_ws, size_t ws_size,
                              hipStream_t stream) {
  (void)in_sizes; (void)n_in; (void)out_size; (void)ws_size;
  const float* u       = (const float*)d_in[0];
  const float* W_in    = (const float*)d_in[1];
  const float* conv_w  = (const float*)d_in[2];
  const float* conv_b  = (const float*)d_in[3];
  const float* dt_bias = (const float*)d_in[4];
  const float* A_log   = (const float*)d_in[5];
  const float* Dvec    = (const float*)d_in[6];
  const float* norm_w  = (const float*)d_in[7];
  const float* W_out   = (const float*)d_in[8];
  float* out = (float*)d_out;

  char* ws = (char*)d_ws;
  size_t off = 0;
  auto alloc = [&](size_t bytes) -> char* {
    char* p = ws + off;
    off += (bytes + 255) & ~(size_t)255;
    return p;
  };
  u16*   u_bf    = (u16*)  alloc((size_t)MTOT * D_MODEL * 2);
  u16*   win_bf  = (u16*)  alloc((size_t)D_IN_PROJ * D_MODEL * 2);
  u16*   wout_bf = (u16*)  alloc((size_t)D_MODEL * D_INNER * 2);
  float* zx      = (float*)alloc((size_t)MTOT * D_IN_PROJ * 4);
  float* xbc     = (float*)alloc((size_t)MTOT * D_XBC * 4);
  float* ybuf    = (float*)alloc((size_t)MTOT * D_INNER * 4);
  float* states  = (float*)alloc((size_t)BATCH * NCHUNK * NHEADS * HEADDIM * D_STATE * 4);
  float* prev    = (float*)alloc((size_t)BATCH * NCHUNK * NHEADS * HEADDIM * D_STATE * 4);
  float* acs     = (float*)alloc((size_t)BATCH * NHEADS * NCHUNK * CHUNK * 4);
  float* cdec    = (float*)alloc((size_t)BATCH * NHEADS * NCHUNK * 4);
  u16*   y_bf    = (u16*)states;   // states dead after scan_kernel; reuse for bf16 y

  // 1) fp32 -> bf16 operand conversion
  {
    int n = MTOT * D_MODEL;
    f2bf_kernel<<<(n + 255) / 256, 256, 0, stream>>>(u, u_bf, n);
  }
  {
    int n = D_IN_PROJ * D_MODEL;
    f2bf_kernel<<<(n + 255) / 256, 256, 0, stream>>>(W_in, win_bf, n);
  }
  {
    int n = D_MODEL * D_INNER;
    f2bf_kernel<<<(n + 255) / 256, 256, 0, stream>>>(W_out, wout_bf, n);
  }

  // 2) in-projection: zxbcdt[4096, 8512] = u . W_in^T   (WMMA, async double-buffer)
  gemm_bf16_abt<<<dim3((D_IN_PROJ + GBN - 1) / GBN, MTOT / GBM), 256, 0, stream>>>(
      u_bf, win_bf, zx, MTOT, D_IN_PROJ, D_MODEL);

  // 3) causal depthwise conv + SiLU on xBC slice
  {
    size_t tot = (size_t)MTOT * D_XBC;
    conv_silu_kernel<<<(unsigned)((tot + 255) / 256), 256, 0, stream>>>(zx, conv_w, conv_b, xbc);
  }

  // 4) per-chunk SSD (Y_diag, states, cumsum) — WMMA heavy, LDS resident
  ssd_chunk_kernel<<<dim3(NCHUNK, NHEADS, BATCH), 256, 0, stream>>>(
      zx, xbc, dt_bias, A_log, ybuf, states, acs, cdec);

  // 5) inter-chunk recurrence
  scan_kernel<<<BATCH * NHEADS, 256, 0, stream>>>(states, cdec, prev);

  // 6) off-diagonal contribution + skip term, accumulated into y  (WMMA)
  ssd_off_kernel<<<dim3(NCHUNK, NHEADS, BATCH), 256, 0, stream>>>(
      xbc, prev, acs, Dvec, ybuf);

  // 7) gated RMSNorm -> bf16
  rmsnorm_gate_kernel<<<MTOT, 256, 0, stream>>>(ybuf, zx, norm_w, y_bf);

  // 8) out-projection: out[4096, 2048] = y . W_out^T   (WMMA, async double-buffer)
  gemm_bf16_abt<<<dim3(D_MODEL / GBN, MTOT / GBM), 256, 0, stream>>>(
      y_bf, wout_bf, out, MTOT, D_MODEL, D_INNER);
}
